// KappaGCNLayer_65901978190372
// MI455X (gfx1250) — matrix operand order repaired and checked by reading, not compiled, not run on hardware
//
#include <hip/hip_runtime.h>
#include <hip/hip_bf16.h>
#include <math.h>

typedef __attribute__((ext_vector_type(16))) __bf16 v16bf;
typedef __attribute__((ext_vector_type(8)))  float  v8f;

#define N_ROWS 8192
#define D_DIM  64
#define NT     80   // 64 nom cols + den col + ones col + 14 pad

// ---------------------------------------------------------------------------
// Kernel 1: XW = mobius_matvec(W, X); emit transposed bf16 panel
//   PT[j][i] = gamma_i * XW[i][j]         (j = 0..63)
//   PT[64][i] = gamma_i - 1
//   PT[65][i] = 1.0                        (row-sum column -> r)
//   PT[66..79][i] = 0                      (pad so N is a multiple of 16)
// ---------------------------------------------------------------------------
__global__ void __launch_bounds__(256)
k1_mobius_panel(const float* __restrict__ X, const float* __restrict__ W,
                __bf16* __restrict__ PT) {
  __shared__ float Ws[D_DIM * D_DIM];
  const int tid = threadIdx.x;
  for (int i = tid; i < D_DIM * D_DIM; i += 256) Ws[i] = W[i];
  __syncthreads();

  const int row = blockIdx.x * 256 + tid;

  float x[D_DIM];
#pragma unroll
  for (int j = 0; j < D_DIM; j += 4) {
    float4 v = *(const float4*)(X + (size_t)row * D_DIM + j);
    x[j] = v.x; x[j + 1] = v.y; x[j + 2] = v.z; x[j + 3] = v.w;
  }

  float nx2 = 0.f;
#pragma unroll
  for (int j = 0; j < D_DIM; j++) nx2 += x[j] * x[j];
  const float nx = sqrtf(fmaxf(nx2, 1e-15f));

  float mx[D_DIM];
#pragma unroll
  for (int j = 0; j < D_DIM; j++) mx[j] = 0.f;
  for (int k = 0; k < D_DIM; k++) {
    const float xk = x[k];
#pragma unroll
    for (int j = 0; j < D_DIM; j++) mx[j] = fmaf(xk, Ws[k * D_DIM + j], mx[j]);
  }

  float nmx2 = 0.f;
#pragma unroll
  for (int j = 0; j < D_DIM; j++) nmx2 += mx[j] * mx[j];
  const float nmx = sqrtf(fmaxf(nmx2, 1e-15f));

  // XW = tanh(nmx/nx * artanh(nx)) * mx / nmx  ->  XW = cfac * mx
  const float art  = atanhf(fminf(fmaxf(nx, -1.f + 1e-7f), 1.f - 1e-7f));
  const float cfac = tanhf((nmx / nx) * art) / nmx;

  // ||XW||^2 = cfac^2 * ||mx||^2 ; gamma = 2 / max(1-||XW||^2, EPS)
  const float xwn2  = cfac * cfac * nmx2;
  const float gamma = 2.f / fmaxf(1.f - xwn2, 1e-7f);
  const float gc    = gamma * cfac;

#pragma unroll
  for (int j = 0; j < D_DIM; j++)
    PT[(size_t)j * N_ROWS + row] = (__bf16)(gc * mx[j]);
  PT[(size_t)64 * N_ROWS + row] = (__bf16)(gamma - 1.f);
  PT[(size_t)65 * N_ROWS + row] = (__bf16)1.f;
#pragma unroll
  for (int j = 66; j < NT; j++)
    PT[(size_t)j * N_ROWS + row] = (__bf16)0.f;
}

// ---------------------------------------------------------------------------
// Kernel 2: [nom | den | r] = A_hat (fp32, cast to bf16 in-regs) @ PT^T
// One wave per 16-row strip of A_hat; 5 N-tiles (80 cols) per A fragment.
// ---------------------------------------------------------------------------
__device__ inline v16bf pack_bf16(float4 a0, float4 a1, float4 a2, float4 a3) {
  v16bf r;
  r[0]  = (__bf16)a0.x; r[1]  = (__bf16)a0.y; r[2]  = (__bf16)a0.z; r[3]  = (__bf16)a0.w;
  r[4]  = (__bf16)a1.x; r[5]  = (__bf16)a1.y; r[6]  = (__bf16)a1.z; r[7]  = (__bf16)a1.w;
  r[8]  = (__bf16)a2.x; r[9]  = (__bf16)a2.y; r[10] = (__bf16)a2.z; r[11] = (__bf16)a2.w;
  r[12] = (__bf16)a3.x; r[13] = (__bf16)a3.y; r[14] = (__bf16)a3.z; r[15] = (__bf16)a3.w;
  return r;
}

__global__ void __launch_bounds__(256)
k2_gyro_gemm(const float* __restrict__ A, const __bf16* __restrict__ PT,
             float* __restrict__ nom, float* __restrict__ den,
             float* __restrict__ rr) {
  const int lane  = threadIdx.x & 31;
  const int wave  = threadIdx.x >> 5;
  const int Mbase = blockIdx.x * 128 + wave * 16;
  const int hi    = lane >> 4;              // half-wave select
  const int lm    = lane & 15;

  const float*  arow  = A + (size_t)(Mbase + lm) * N_ROWS;
  const int     koffA = hi * 16;            // A frag: lanes16-31 hold K=k+16..k+31
  const int     koffB = hi * 16;            // B frag: lanes16-31 hold K=k+16..k+31

  const __bf16* b0p = PT + (size_t)(0 * 16 + lm) * N_ROWS + koffB;
  const __bf16* b1p = PT + (size_t)(1 * 16 + lm) * N_ROWS + koffB;
  const __bf16* b2p = PT + (size_t)(2 * 16 + lm) * N_ROWS + koffB;
  const __bf16* b3p = PT + (size_t)(3 * 16 + lm) * N_ROWS + koffB;
  const __bf16* b4p = PT + (size_t)(4 * 16 + lm) * N_ROWS + koffB;

  v8f c0 = {}, c1 = {}, c2 = {}, c3 = {}, c4 = {};

  for (int k = 0; k < N_ROWS; k += 32) {
    // A fragment: 16x32 bf16 (lane = M, elements 0..15: K=koffA..koffA+15)
    float4 a0 = *(const float4*)(arow + k + koffA);
    float4 a1 = *(const float4*)(arow + k + koffA + 4);
    float4 a2 = *(const float4*)(arow + k + koffA + 8);
    float4 a3 = *(const float4*)(arow + k + koffA + 12);
    v16bf af = pack_bf16(a0, a1, a2, a3);

    // B fragments: contiguous 16 bf16 per lane from the pre-transposed panel
    v16bf b0 = *(const v16bf*)(b0p + k);
    v16bf b1 = *(const v16bf*)(b1p + k);
    v16bf b2 = *(const v16bf*)(b2p + k);
    v16bf b3 = *(const v16bf*)(b3p + k);
    v16bf b4 = *(const v16bf*)(b4p + k);

    c0 = __builtin_amdgcn_wmma_f32_16x16x32_bf16(false, af, false, b0, (short)0, c0, false, false);
    c1 = __builtin_amdgcn_wmma_f32_16x16x32_bf16(false, af, false, b1, (short)0, c1, false, false);
    c2 = __builtin_amdgcn_wmma_f32_16x16x32_bf16(false, af, false, b2, (short)0, c2, false, false);
    c3 = __builtin_amdgcn_wmma_f32_16x16x32_bf16(false, af, false, b3, (short)0, c3, false, false);
    c4 = __builtin_amdgcn_wmma_f32_16x16x32_bf16(false, af, false, b4, (short)0, c4, false, false);
  }

  // D layout: VGPR j -> M = Mbase + j + hi*8 ; lane%16 -> N
#pragma unroll
  for (int j = 0; j < 8; j++) {
    const int m = Mbase + j + hi * 8;
    float* nr = nom + (size_t)m * D_DIM;
    nr[0  + lm] = c0[j];
    nr[16 + lm] = c1[j];
    nr[32 + lm] = c2[j];
    nr[48 + lm] = c3[j];
    if (lm == 0)      den[m] = c4[j];   // column 64: A_hat @ (gamma-1)
    else if (lm == 1) rr[m]  = c4[j];   // column 65: A_hat @ 1  (row sums)
  }
}

// ---------------------------------------------------------------------------
// Kernel 3: per-row epilogue (gyromidpoint halving, mobius scalar mul,
// expmap0(relu(logmap0()))) — one wave per row, 2 cols per lane.
// ---------------------------------------------------------------------------
__device__ inline float wsum32(float v) {
#pragma unroll
  for (int o = 16; o > 0; o >>= 1) v += __shfl_xor(v, o, 32);
  return v;
}

__global__ void __launch_bounds__(256)
k3_epilogue(const float* __restrict__ nom, const float* __restrict__ den,
            const float* __restrict__ rr, float* __restrict__ out) {
  const int lane = threadIdx.x & 31;
  const int wave = threadIdx.x >> 5;
  const int row  = blockIdx.x * 8 + wave;

  const float* nr = nom + (size_t)row * D_DIM;
  float d = den[row];
  d = (fabsf(d) < 1e-10f) ? 1e-10f : d;

  float t0 = nr[lane] / d;
  float t1 = nr[lane + 32] / d;

  // halve: t / (1 + sqrt(1 - ||t||^2))
  const float sq = wsum32(t0 * t0 + t1 * t1);
  const float s  = 1.f / (1.f + sqrtf(fmaxf(1.f - sq, 0.f)));
  t0 *= s; t1 *= s;

  // mobius_scalar_mul(r, mid)
  const float n  = sqrtf(fmaxf(wsum32(t0 * t0 + t1 * t1), 1e-15f));
  const float r  = rr[row];
  const float f  = tanhf(r * atanhf(fminf(n, 1.f - 1e-7f))) / n;
  t0 *= f; t1 *= f;

  // logmap0
  const float nA = sqrtf(fmaxf(wsum32(t0 * t0 + t1 * t1), 1e-15f));
  const float g  = atanhf(fminf(nA, 1.f - 1e-7f)) / nA;
  t0 = fmaxf(t0 * g, 0.f);           // relu
  t1 = fmaxf(t1 * g, 0.f);

  // expmap0
  const float nu = sqrtf(fmaxf(wsum32(t0 * t0 + t1 * t1), 1e-15f));
  const float h  = tanhf(nu) / nu;

  out[(size_t)row * D_DIM + lane]      = t0 * h;
  out[(size_t)row * D_DIM + lane + 32] = t1 * h;
}

// ---------------------------------------------------------------------------
extern "C" void kernel_launch(void* const* d_in, const int* in_sizes, int n_in,
                              void* d_out, int out_size, void* d_ws, size_t ws_size,
                              hipStream_t stream) {
  const float* X = (const float*)d_in[0];   // [8192, 64]
  const float* A = (const float*)d_in[1];   // [8192, 8192]
  const float* W = (const float*)d_in[2];   // [64, 64]
  float* out = (float*)d_out;               // [8192, 64]

  char* ws = (char*)d_ws;
  __bf16* PT  = (__bf16*)ws;                                        // 80*8192 bf16
  float*  nom = (float*)(ws + (size_t)NT * N_ROWS * 2);             // 8192*64 f32
  float*  den = (float*)(ws + (size_t)NT * N_ROWS * 2
                            + (size_t)N_ROWS * D_DIM * 4);          // 8192 f32
  float*  rr  = den + N_ROWS;                                       // 8192 f32

  k1_mobius_panel<<<N_ROWS / 256, 256, 0, stream>>>(X, W, PT);
  k2_gyro_gemm<<<N_ROWS / 128, 256, 0, stream>>>(A, PT, nom, den, rr);
  k3_epilogue<<<N_ROWS / 8, 256, 0, stream>>>(nom, den, rr, out);
}
